// MotionAGTransformerDecoderLayer_6339371729451
// MI455X (gfx1250) — compile-verified, matrix operand order checked by reading, not run on hardware
//
#include <hip/hip_runtime.h>
#include <math.h>

typedef _Float16 f16;
typedef __attribute__((ext_vector_type(16))) _Float16 v16h;
typedef __attribute__((ext_vector_type(8)))  _Float16 v8h;
typedef __attribute__((ext_vector_type(8)))  float    v8f;

#define TT 243
#define PP 17
#define LL 4
#define CC 128
#define HH 8
#define BB 2
#define DHH 16
#define M1_ (BB*TT*PP*LL)        // 33048
#define MP1_ 33056               // padded: /16 and /32
#define M2_ (BB*TT*PP*(LL+1))    // 41310
#define MP2_ 41312               // padded: /16 and /32

// ---- fragment loader -------------------------------------------------------
// CDNA5 16-bit A/B fragment per lane: K pattern = kh + {0..7} and kh + 16 + {0..7}
// where kh = (lane>>4)*8.  Two contiguous 16-byte loads per lane.
__device__ __forceinline__ v16h ld_frag(const f16* p) {
  v8h lo = *(const v8h*)(p);
  v8h hi = *(const v8h*)(p + 16);
  v16h r;
#pragma unroll
  for (int i = 0; i < 8; ++i) { r[i] = lo[i]; r[i + 8] = hi[i]; }
  return r;
}

// ---- small utility kernels -------------------------------------------------
__global__ void cast_f16_kernel(const float* __restrict__ s, f16* __restrict__ d, int n) {
  int i = blockIdx.x * 256 + threadIdx.x;
  if (i < n) d[i] = (f16)s[i];
}

__global__ void zero2_kernel(f16* p0, int n0, f16* p1, int n1) {
  int t = blockIdx.x * 256 + threadIdx.x;
  if (t < n0) p0[t] = (f16)0.f;
  if (t < n1) p1[t] = (f16)0.f;
}

// copy x[..., 0, :] (slot 0) into xcat (same flat layout, rows of 5*128 floats)
__global__ void copy_x0_kernel(const float* __restrict__ x, float* __restrict__ xcat) {
  int t = blockIdx.x * 256 + threadIdx.x;
  if (t >= BB * TT * PP * 128) return;
  int g = t >> 7, c = t & 127;
  xcat[(long)g * 640 + c] = x[(long)g * 640 + c];
}

// ---- LayerNorm -------------------------------------------------------------
// mode 0: src = x, row remapped (r -> slot 1+r%4 of group r/4); writes xn (out)
//         and q = xn + pe + learn_pos (out2).  mode 1: plain LN on src rows.
// Pad rows (r >= M) write zeros so WMMA A-fragments never need guarding.
__global__ void ln_kernel(const float* __restrict__ src, const float* __restrict__ g,
                          const float* __restrict__ be, f16* __restrict__ out,
                          f16* __restrict__ out2, const float* __restrict__ pe,
                          const float* __restrict__ lp, int M, int mode) {
  int r = blockIdx.x;
  int c = threadIdx.x;
  __shared__ float red[128];
  float v = 0.f;
  if (r < M) {
    long srow = (mode == 0) ? ((long)(r >> 2) * 5 + 1 + (r & 3)) : (long)r;
    v = src[srow * 128 + c];
  }
  red[c] = v; __syncthreads();
  for (int s = 64; s > 0; s >>= 1) { if (c < s) red[c] += red[c + s]; __syncthreads(); }
  float mean = red[0] * (1.f / 128.f);
  __syncthreads();
  float d = v - mean;
  red[c] = d * d; __syncthreads();
  for (int s = 64; s > 0; s >>= 1) { if (c < s) red[c] += red[c + s]; __syncthreads(); }
  float var = red[0] * (1.f / 128.f);
  float y = (r < M) ? (d * rsqrtf(var + 1e-5f) * g[c] + be[c]) : 0.f;
  out[(long)r * 128 + c] = (f16)y;
  if (mode == 0) {
    float q = 0.f;
    if (r < M) {
      int pr = r % (TT * PP * LL);   // strip batch: pe/learn_pos are (1,T,P,L,C)
      q = y + pe[(long)pr * 128 + c] + lp[(long)pr * 128 + c];
    }
    out2[(long)r * 128 + c] = (f16)q;
  }
}

// ---- GEMM epilogue (shared by the four 16x16 sub-tiles of a wave) ----------
//   0: fp32 store [r*N+n], optional residual (same index)
//   1: remap r -> 5-slot row, residual from res (x layout), store to outF (xcat)
//   2: scatter into v_maps[l][b][h][t][p][dh]
//   3: exact-erf GELU, f16 store (pad rows store 0 so buffer is WMMA-A safe)
__device__ __forceinline__ void gemm_epilogue(v8f acc, int m0, int n0, int lane,
                                              const float* __restrict__ bias,
                                              float* __restrict__ outF,
                                              f16* __restrict__ outH,
                                              const float* __restrict__ res,
                                              int M, int N, int mode) {
  int lm = lane & 15;
  int n = n0 + lm;
  float bn = bias[n];
  int rb = m0 + ((lane >> 4) << 3);          // C/D layout: lanes 16-31 hold rows+8
#pragma unroll
  for (int i = 0; i < 8; ++i) {
    int r = rb + i;
    float y = acc[i] + bn;
    if (mode == 3) {
      float gl = (r < M) ? 0.5f * y * (1.f + erff(y * 0.70710678118f)) : 0.f;
      outH[(long)r * N + n] = (f16)gl;
    } else if (r < M) {
      if (mode == 0) {
        long o = (long)r * N + n;
        outF[o] = res ? (y + res[o]) : y;
      } else if (mode == 1) {
        long o = ((long)(r >> 2) * 5 + 1 + (r & 3)) * 128 + n;
        outF[o] = y + res[o];
      } else {  // mode 2: v_maps scatter
        int l = r & 3;
        int rq = r >> 2;
        int pp = rq % PP;
        int rq2 = rq / PP;
        int tt = rq2 % TT;
        int b = rq2 / TT;
        int hh = n >> 4, dd = n & 15;
        long o = (((((long)l * BB + b) * HH + hh) * TT + tt) * PP + pp) * DHH + dd;
        outF[o] = y;
      }
    }
  }
}

// ---- generic WMMA GEMM: Y = A(f16, MPxK) @ W(f16, NxK)^T + bias ------------
// 2x2 register-blocked: one 32x32 output tile per wave -> 4 fragment loads
// feed 4 WMMAs per k-step (1:1 load:wmma), A/B fragments reused twice each.
__global__ void gemm_kernel(const f16* __restrict__ A, const f16* __restrict__ W,
                            const float* __restrict__ bias, float* __restrict__ outF,
                            f16* __restrict__ outH, const float* __restrict__ res,
                            int M, int MP, int N, int K, int mode) {
  int wave = (blockIdx.x * 256 + threadIdx.x) >> 5;
  int lane = threadIdx.x & 31;
  int ntN = N >> 5;
  int total = (MP >> 5) * ntN;
  if (wave >= total) return;                 // wave-uniform exit, EXEC stays full
  int mt = wave / ntN, nt = wave - mt * ntN;
  int m0 = mt << 5, n0 = nt << 5;
  int lm = lane & 15;
  int kh = (lane >> 4) << 3;
  const f16* a0 = A + (long)(m0 + lm) * K + kh;
  const f16* a1 = a0 + (long)16 * K;
  const f16* b0 = W + (long)(n0 + lm) * K + kh;
  const f16* b1 = b0 + (long)16 * K;
  v8f acc00 = {0.f, 0.f, 0.f, 0.f, 0.f, 0.f, 0.f, 0.f};
  v8f acc01 = acc00, acc10 = acc00, acc11 = acc00;
  for (int k0 = 0; k0 < K; k0 += 32) {
    v16h fa0 = ld_frag(a0 + k0);
    v16h fa1 = ld_frag(a1 + k0);
    v16h fb0 = ld_frag(b0 + k0);
    v16h fb1 = ld_frag(b1 + k0);
    acc00 = __builtin_amdgcn_wmma_f32_16x16x32_f16(false, fa0, false, fb0, (short)0,
                                                   acc00, false, false);
    acc01 = __builtin_amdgcn_wmma_f32_16x16x32_f16(false, fa0, false, fb1, (short)0,
                                                   acc01, false, false);
    acc10 = __builtin_amdgcn_wmma_f32_16x16x32_f16(false, fa1, false, fb0, (short)0,
                                                   acc10, false, false);
    acc11 = __builtin_amdgcn_wmma_f32_16x16x32_f16(false, fa1, false, fb1, (short)0,
                                                   acc11, false, false);
  }
  gemm_epilogue(acc00, m0,      n0,      lane, bias, outF, outH, res, M, N, mode);
  gemm_epilogue(acc01, m0,      n0 + 16, lane, bias, outF, outH, res, M, N, mode);
  gemm_epilogue(acc10, m0 + 16, n0,      lane, bias, outF, outH, res, M, N, mode);
  gemm_epilogue(acc11, m0 + 16, n0 + 16, lane, bias, outF, outH, res, M, N, mode);
}

// ---- softmax over (L*NPTS)=16 attention weights per (row, head) ------------
__global__ void softmax16_kernel(float* __restrict__ aw, int m) {
  int t = blockIdx.x * 256 + threadIdx.x;
  if (t >= m * 8) return;
  float* p = aw + (long)(t >> 3) * 128 + (t & 7) * 16;
  float v[16];
  float mx = -1e30f;
#pragma unroll
  for (int j = 0; j < 16; ++j) { v[j] = p[j]; mx = fmaxf(mx, v[j]); }
  float s = 0.f;
#pragma unroll
  for (int j = 0; j < 16; ++j) { v[j] = __expf(v[j] - mx); s += v[j]; }
  float inv = 1.f / s;
#pragma unroll
  for (int j = 0; j < 16; ++j) p[j] = v[j] * inv;
}

// ---- deformable bilinear sampling ------------------------------------------
// one thread per (row, h, dh); off/aw fp32, v_maps fp32, output f16 (GEMM A).
__global__ void sample_kernel(const float* __restrict__ off, const float* __restrict__ aw,
                              const float* __restrict__ vm, const float* __restrict__ refp,
                              f16* __restrict__ o16) {
  long tid = (long)blockIdx.x * 256 + threadIdx.x;
  if (tid >= (long)M1_ * 128) return;
  int row = (int)(tid >> 7);
  int n = (int)(tid & 127);
  int h = n >> 4, dh = n & 15;
  int rq = row >> 2;                 // (b*T+t)*P + p
  float rx = refp[(long)rq * 2 + 0];
  float ry = refp[(long)rq * 2 + 1];
  int b = rq / (TT * PP);
  const float* offr = off + (long)row * 256 + h * 32;
  const float* awr = aw + (long)row * 128 + h * 16;
  float acc = 0.f;
  for (int lvl = 0; lvl < 4; ++lvl) {
    const float* vmb = vm + ((((long)lvl * BB + b) * HH + h) * TT) * PP * DHH;
#pragma unroll
    for (int pt = 0; pt < 4; ++pt) {
      float ox = offr[lvl * 8 + pt * 2];
      float oy = offr[lvl * 8 + pt * 2 + 1];
      // x = ((ref + off/norm)*2-1 +1)*W/2 - 0.5  ==  ref*W + off - 0.5
      float xf = rx * (float)PP + ox - 0.5f;
      float yf = ry * (float)TT + oy - 0.5f;
      float x0f = floorf(xf), y0f = floorf(yf);
      int x0 = (int)x0f, y0 = (int)y0f;
      float wx1 = xf - x0f, wx0 = 1.f - wx1;
      float wy1 = yf - y0f, wy0 = 1.f - wy1;
      float s = 0.f;
#pragma unroll
      for (int cy = 0; cy < 2; ++cy) {
#pragma unroll
        for (int cx = 0; cx < 2; ++cx) {
          int yi = y0 + cy, xi = x0 + cx;
          bool ok = (yi >= 0) && (yi < TT) && (xi >= 0) && (xi < PP);
          float v = ok ? vmb[((long)yi * PP + xi) * DHH + dh] : 0.f;
          s += v * (cy ? wy1 : wy0) * (cx ? wx1 : wx0);
        }
      }
      acc += s * awr[lvl * 4 + pt];
    }
  }
  o16[(long)row * 128 + n] = (f16)acc;
}

// ---- self-attention per (b*t, h): S=85 (pad 96), DH=16 (pad K to 32) -------
__global__ void attn_kernel(const float* __restrict__ qkv, f16* __restrict__ ca16) {
  __shared__ __align__(16) f16 sQ[96 * 32];
  __shared__ __align__(16) f16 sK[96 * 32];
  __shared__ __align__(16) f16 sVT[16 * 96];
  __shared__ __align__(16) f16 sA[96 * 96];
  int bt = blockIdx.x >> 3;
  int h = blockIdx.x & 7;
  int tid = threadIdx.x;
  for (int i = tid; i < 96 * 32; i += 128) { sQ[i] = (f16)0.f; sK[i] = (f16)0.f; }
  for (int i = tid; i < 16 * 96; i += 128) sVT[i] = (f16)0.f;
  __syncthreads();
  const float* base = qkv + (long)bt * 85 * 384 + h * 16;
  for (int i = tid; i < 85 * 16; i += 128) {
    int s = i >> 4, d = i & 15;
    const float* rp = base + (long)s * 384 + d;
    sQ[s * 32 + d] = (f16)rp[0];     // q
    sK[s * 32 + d] = (f16)rp[128];   // k
    sVT[d * 96 + s] = (f16)rp[256];  // v, transposed for B-fragments
  }
  __syncthreads();
  int lane = tid & 31;
  int wv = tid >> 5;
  int lm = lane & 15;
  int kh = (lane >> 4) << 3;
  // scores = q @ k^T * (1/sqrt(16)); 36 16x16 tiles over 4 waves
  for (int t = wv; t < 36; t += 4) {
    int m0 = (t / 6) << 4, n0 = (t % 6) << 4;
    v8f acc = {0.f, 0.f, 0.f, 0.f, 0.f, 0.f, 0.f, 0.f};
    v16h a = ld_frag(&sQ[(m0 + lm) * 32 + kh]);
    v16h b = ld_frag(&sK[(n0 + lm) * 32 + kh]);
    acc = __builtin_amdgcn_wmma_f32_16x16x32_f16(false, a, false, b, (short)0, acc,
                                                 false, false);
    int rb = m0 + ((lane >> 4) << 3);
#pragma unroll
    for (int i = 0; i < 8; ++i) sA[(rb + i) * 96 + n0 + lm] = (f16)(acc[i] * 0.25f);
  }
  __syncthreads();
  // fp32 softmax over 85 valid columns; pad rows/cols -> 0
  for (int row = tid; row < 96; row += 128) {
    if (row < 85) {
      float mx = -1e30f;
      for (int j = 0; j < 85; ++j) mx = fmaxf(mx, (float)sA[row * 96 + j]);
      float s = 0.f;
      for (int j = 0; j < 85; ++j) s += __expf((float)sA[row * 96 + j] - mx);
      float inv = 1.f / s;
      for (int j = 0; j < 85; ++j)
        sA[row * 96 + j] = (f16)(__expf((float)sA[row * 96 + j] - mx) * inv);
      for (int j = 85; j < 96; ++j) sA[row * 96 + j] = (f16)0.f;
    } else {
      for (int j = 0; j < 96; ++j) sA[row * 96 + j] = (f16)0.f;
    }
  }
  __syncthreads();
  // ca = att @ v : 6 M-tiles x 1 N-tile, K = 96 (3 WMMA steps)
  for (int t = wv; t < 6; t += 4) {
    int m0 = t << 4;
    v8f acc = {0.f, 0.f, 0.f, 0.f, 0.f, 0.f, 0.f, 0.f};
    for (int k0 = 0; k0 < 96; k0 += 32) {
      v16h a = ld_frag(&sA[(m0 + lm) * 96 + k0 + kh]);
      v16h b = ld_frag(&sVT[lm * 96 + k0 + kh]);
      acc = __builtin_amdgcn_wmma_f32_16x16x32_f16(false, a, false, b, (short)0, acc,
                                                   false, false);
    }
    int rb = m0 + ((lane >> 4) << 3);
#pragma unroll
    for (int i = 0; i < 8; ++i) {
      int r = rb + i;
      if (r < 85) ca16[((long)bt * 85 + r) * 128 + h * 16 + lm] = (f16)acc[i];
    }
  }
}

// ---- host orchestration ----------------------------------------------------
extern "C" void kernel_launch(void* const* d_in, const int* in_sizes, int n_in,
                              void* d_out, int out_size, void* d_ws, size_t ws_size,
                              hipStream_t stream) {
  (void)in_sizes; (void)n_in; (void)out_size; (void)ws_size;
  const float* X    = (const float*)d_in[0];
  const float* REF  = (const float*)d_in[1];
  const float* PE   = (const float*)d_in[2];
  const float* LP   = (const float*)d_in[3];
  const float* Woff = (const float*)d_in[4];  const float* Boff = (const float*)d_in[5];
  const float* Wat  = (const float*)d_in[6];  const float* Bat  = (const float*)d_in[7];
  const float* Wval = (const float*)d_in[8];  const float* Bval = (const float*)d_in[9];
  const float* Wout = (const float*)d_in[10]; const float* Bout = (const float*)d_in[11];
  const float* Win  = (const float*)d_in[12]; const float* Bin  = (const float*)d_in[13];
  const float* Wow  = (const float*)d_in[14]; const float* Bow  = (const float*)d_in[15];
  const float* Wf1  = (const float*)d_in[16]; const float* Bf1  = (const float*)d_in[17];
  const float* Wf2  = (const float*)d_in[18]; const float* Bf2  = (const float*)d_in[19];
  const float* G1 = (const float*)d_in[20]; const float* Be1 = (const float*)d_in[21];
  const float* G2 = (const float*)d_in[22]; const float* Be2 = (const float*)d_in[23];
  const float* G3 = (const float*)d_in[24]; const float* Be3 = (const float*)d_in[25];
  float* OUT = (float*)d_out;

  char* ws = (char*)d_ws;
  size_t cur = 0;
  auto alloc = [&](size_t bytes) -> char* {
    char* p = ws + cur;
    cur += (bytes + 255) & ~(size_t)255;
    return p;
  };
  f16* w16_off = (f16*)alloc((size_t)256 * 128 * 2);
  f16* w16_at  = (f16*)alloc((size_t)128 * 128 * 2);
  f16* w16_val = (f16*)alloc((size_t)128 * 128 * 2);
  f16* w16_out = (f16*)alloc((size_t)128 * 128 * 2);
  f16* w16_in  = (f16*)alloc((size_t)384 * 128 * 2);
  f16* w16_ow  = (f16*)alloc((size_t)128 * 128 * 2);
  f16* w16_f1  = (f16*)alloc((size_t)512 * 128 * 2);
  f16* w16_f2  = (f16*)alloc((size_t)128 * 512 * 2);
  f16* q16  = (f16*)alloc((size_t)MP1_ * 128 * 2);
  f16* xn16 = (f16*)alloc((size_t)MP1_ * 128 * 2);
  // BIG region: phase-A {off, aw, v_maps} -> phase-B qkv -> phase-C gelu(f16)
  size_t offSz = (size_t)MP1_ * 256 * 4;
  size_t awSz  = (size_t)MP1_ * 128 * 4;
  size_t vmSz  = (size_t)LL * BB * HH * TT * PP * DHH * 4;
  size_t bigSz = offSz + awSz + vmSz;
  size_t qkvSz = (size_t)MP2_ * 384 * 4;
  size_t g16Sz = (size_t)MP2_ * 512 * 2;
  if (qkvSz > bigSz) bigSz = qkvSz;
  if (g16Sz > bigSz) bigSz = g16Sz;
  char* big = alloc(bigSz);
  float* offb = (float*)big;
  float* awb  = (float*)(big + offSz);
  float* vmb  = (float*)(big + offSz + awSz);
  float* qkvb = (float*)big;
  f16*   g16  = (f16*)big;
  f16* o16    = (f16*)alloc((size_t)MP1_ * 128 * 2);
  float* xcat = (float*)alloc((size_t)MP2_ * 128 * 4);
  f16* xn2h   = (f16*)alloc((size_t)MP2_ * 128 * 2);  // xn2, later reused as h
  f16* ca16   = (f16*)alloc((size_t)MP2_ * 128 * 2);
  float* xcat2 = (float*)alloc((size_t)MP2_ * 128 * 4);

  auto castN = [&](const float* s, f16* d, int n) {
    cast_f16_kernel<<<(n + 255) / 256, 256, 0, stream>>>(s, d, n);
  };
  castN(Woff, w16_off, 256 * 128);
  castN(Wat,  w16_at,  128 * 128);
  castN(Wval, w16_val, 128 * 128);
  castN(Wout, w16_out, 128 * 128);
  castN(Win,  w16_in,  384 * 128);
  castN(Wow,  w16_ow,  128 * 128);
  castN(Wf1,  w16_f1,  512 * 128);
  castN(Wf2,  w16_f2,  128 * 512);
  zero2_kernel<<<4, 256, 0, stream>>>(o16 + (size_t)M1_ * 128, (MP1_ - M1_) * 128,
                                      ca16 + (size_t)M2_ * 128, (MP2_ - M2_) * 128);
  copy_x0_kernel<<<(BB * TT * PP * 128 + 255) / 256, 256, 0, stream>>>(X, xcat);

  auto gemm = [&](const f16* A, const f16* W, const float* bias, float* oF, f16* oH,
                  const float* res, int M, int MP, int N, int K, int mode) {
    int waves = (MP >> 5) * (N >> 5);        // one 32x32 tile per wave
    gemm_kernel<<<(waves + 7) / 8, 256, 0, stream>>>(A, W, bias, oF, oH, res,
                                                     M, MP, N, K, mode);
  };

  // 1) LN1 + positional add -> xn16, q16
  ln_kernel<<<MP1_, 128, 0, stream>>>(X, G1, Be1, xn16, q16, PE, LP, M1_, 0);
  // 2) offsets / attention logits / values
  gemm(q16,  w16_off, Boff, offb, nullptr, nullptr, M1_, MP1_, 256, 128, 0);
  gemm(q16,  w16_at,  Bat,  awb,  nullptr, nullptr, M1_, MP1_, 128, 128, 0);
  softmax16_kernel<<<(M1_ * 8 + 255) / 256, 256, 0, stream>>>(awb, M1_);
  gemm(xn16, w16_val, Bval, vmb,  nullptr, nullptr, M1_, MP1_, 128, 128, 2);
  // 3) deformable bilinear sampling -> o16
  {
    long nthr = (long)M1_ * 128;
    sample_kernel<<<(int)((nthr + 255) / 256), 256, 0, stream>>>(offb, awb, vmb, REF, o16);
  }
  // 4) output projection + residual(x), scattered into xcat slots 1..4
  gemm(o16, w16_out, Bout, xcat, nullptr, X, M1_, MP1_, 128, 128, 1);
  // 5) LN2 -> xn2, qkv projection
  ln_kernel<<<MP2_, 128, 0, stream>>>(xcat, G2, Be2, xn2h, nullptr, nullptr, nullptr, M2_, 1);
  gemm(xn2h, w16_in, Bin, qkvb, nullptr, nullptr, M2_, MP2_, 384, 128, 0);
  // 6) self-attention -> ca16
  attn_kernel<<<BB * TT * HH, 128, 0, stream>>>(qkvb, ca16);
  // 7) attention out projection + residual(xcat) -> xcat2
  gemm(ca16, w16_ow, Bow, xcat2, nullptr, xcat, M2_, MP2_, 128, 128, 0);
  // 8) LN3 -> h ; FFN fc1(+GELU, f16) ; fc2 + residual(xcat2) -> d_out
  ln_kernel<<<MP2_, 128, 0, stream>>>(xcat2, G3, Be3, xn2h, nullptr, nullptr, nullptr, M2_, 1);
  gemm(xn2h, w16_f1, Bf1, nullptr, g16, nullptr, M2_, MP2_, 512, 128, 3);
  gemm(g16,  w16_f2, Bf2, OUT, nullptr, xcat2, M2_, MP2_, 128, 512, 0);
}
